// HORN_88768384074041
// MI455X (gfx1250) — matrix operand order repaired and checked by reading, not compiled, not run on hardware
//
#include <hip/hip_runtime.h>
#include <math.h>

typedef __attribute__((ext_vector_type(2)))  float     v2f;
typedef __attribute__((ext_vector_type(8)))  float     v8f;
typedef __attribute__((ext_vector_type(16))) _Float16  v16h;
typedef __attribute__((ext_vector_type(8)))  _Float16  v8h;

#define T_STEPS 512
#define BATCH   256
#define I_DIM   128
#define N_DIM   1024
#define O_DIM   10

#define H_STEP_F    0.1f
#define OMEGA_F     1.0f     // OMEGA^2
#define GAMMA2_F    0.2f     // 2*GAMMA
#define GAIN_REC_F  0.03125f // 1/sqrt(1024)

// ---------------- init: zero x, y (f32) and y16 ping buffer ----------------
__global__ __launch_bounds__(256) void horn_init(float* __restrict__ x,
                                                 float* __restrict__ y,
                                                 _Float16* __restrict__ y16) {
  int i = blockIdx.x * blockDim.x + threadIdx.x;
  if (i < BATCH * N_DIM) { x[i] = 0.0f; y[i] = 0.0f; y16[i] = (_Float16)0.0f; }
}

// ---------------- convert h2h_w (f32 [N,N]) -> f16 once per call ----------------
__global__ __launch_bounds__(256) void horn_cvt_w(const float* __restrict__ w,
                                                  _Float16* __restrict__ w16) {
  int i = blockIdx.x * blockDim.x + threadIdx.x;
  if (i < N_DIM * N_DIM) w16[i] = (_Float16)w[i];
}

// ---------------- one recurrent step (f16-XDL rec GEMM + f32 in-proj + pointwise) ----
// pre[b,n] = (batch_t @ i2h_w^T)[b,n] + i2h_b[n]
//          + GAIN_REC * ((y16 @ h16^T)[b,n] + h2h_b[n])
// y' = y + h*(tanh(pre) - w^2*x - 2g*y);  x' = x + h*y'
__global__ __launch_bounds__(128) void horn_step(
    const float*    __restrict__ batch_t,  // [B, I] slice for this t (f32)
    const float*    __restrict__ i2h_w,    // [N, I] (f32)
    const float*    __restrict__ i2h_b,    // [N]
    const _Float16* __restrict__ h16,      // [N, N] h2h_w in f16
    const float*    __restrict__ h2h_b,    // [N]
    float*          __restrict__ x,        // [B, N] in-place
    float*          __restrict__ y,        // [B, N] in-place (own-element only)
    const _Float16* __restrict__ y16_in,   // [B, N] f16 copy of y (read by all)
    _Float16*       __restrict__ y16_out)  // [B, N] f16 copy for next step
{
  const int lane = threadIdx.x & 31;
  const int wave = threadIdx.x >> 5;
  // 16 M-tiles x 64 N-tiles of 16x16; 4 waves/block -> 256 blocks
  const int mt = blockIdx.x >> 4;        // 0..15
  const int ng = blockIdx.x & 15;        // 0..15
  const int b0 = mt * 16;
  const int n0 = ng * 64 + wave * 16;

  const int mrow = lane & 15;
  const int ncol = n0 + (lane & 15);
  const int hi   = lane >> 4;            // 0 or 1

  // ---- recurrent GEMM in f16 (XDL path): C_rec += y16[16x32] * h16^T[32x16]
  // A 16x32 f16 per lane: K chunk  [ka, ka+16), ka = hi*16 (two v8h)
  // B 32x16 f16 per lane: K chunk  [kb, kb+16), kb = hi*16  (contiguous v16h)
  const _Float16* pa = y16_in + (size_t)(b0 + mrow) * N_DIM + hi * 16;
  const _Float16* pb = h16    + (size_t)ncol        * N_DIM + hi * 16;
  v8f acc_rec = {};
  #pragma unroll 4
  for (int k = 0; k < N_DIM; k += 32) {
    union { v16h v; v8h h[2]; } a;
    a.h[0] = *(const v8h*)(pa + k);
    a.h[1] = *(const v8h*)(pa + k + 8);
    v16h b = *(const v16h*)(pb + k);
    acc_rec = __builtin_amdgcn_wmma_f32_16x16x32_f16(
        false, a.v, false, b, (short)0, acc_rec, false, false);
  }

  // ---- input projection in exact f32 WMMA: C_in += batch_t[16x4] * i2h_w^T[4x16]
  const int kp = hi * 2;                 // f32 A/B: K pair 0-1 or 2-3
  const float* urow = batch_t + (size_t)(b0 + mrow) * I_DIM + kp;
  const float* wrow = i2h_w   + (size_t)ncol        * I_DIM + kp;
  v8f acc_in = {};
  #pragma unroll 4
  for (int k = 0; k < I_DIM; k += 4) {
    v2f a = *(const v2f*)(urow + k);
    v2f b = *(const v2f*)(wrow + k);
    acc_in = __builtin_amdgcn_wmma_f32_16x16x4_f32(
        false, a, false, b, (short)0, acc_in, false, false);
  }

  // ---- epilogue: bias, gain, tanh, Euler update (all f32)
  const float bias  = i2h_b[ncol] + GAIN_REC_F * h2h_b[ncol];
  const int   mbase = b0 + (hi << 3);    // C/D: VGPR r -> row r (+8 for hi half)
  #pragma unroll
  for (int r = 0; r < 8; ++r) {
    const int idx = (mbase + r) * N_DIM + ncol;
    float pre = acc_in[r] + GAIN_REC_F * acc_rec[r] + bias;
    float yo  = y[idx];
    float xo  = x[idx];
    float yn  = yo + H_STEP_F * (tanhf(pre) - OMEGA_F * xo - GAMMA2_F * yo);
    float xn  = xo + H_STEP_F * yn;
    y[idx]      = yn;
    x[idx]      = xn;
    y16_out[idx] = (_Float16)yn;
  }
}

// ---------------- readout: out = x_T @ h2o_w^T + h2o_b  (tiny: [256,10]) ----------------
__global__ __launch_bounds__(256) void horn_out(const float* __restrict__ x,
                                                const float* __restrict__ h2o_w,
                                                const float* __restrict__ h2o_b,
                                                float* __restrict__ out) {
  int i = blockIdx.x * blockDim.x + threadIdx.x;
  if (i >= BATCH * O_DIM) return;
  int b = i / O_DIM, o = i % O_DIM;
  const float* xr = x + (size_t)b * N_DIM;
  const float* wr = h2o_w + (size_t)o * N_DIM;
  float s = h2o_b[o];
  #pragma unroll 8
  for (int n = 0; n < N_DIM; ++n) s += xr[n] * wr[n];
  out[i] = s;
}

extern "C" void kernel_launch(void* const* d_in, const int* in_sizes, int n_in,
                              void* d_out, int out_size, void* d_ws, size_t ws_size,
                              hipStream_t stream) {
  (void)in_sizes; (void)n_in; (void)out_size; (void)ws_size;

  const float* batch  = (const float*)d_in[0];   // [T, B, I]
  const float* i2h_w  = (const float*)d_in[1];   // [N, I]
  const float* i2h_b  = (const float*)d_in[2];   // [N]
  const float* h2h_w  = (const float*)d_in[3];   // [N, N]
  const float* h2h_b  = (const float*)d_in[4];   // [N]
  const float* h2o_w  = (const float*)d_in[5];   // [O, N]
  const float* h2o_b  = (const float*)d_in[6];   // [O]
  float* out = (float*)d_out;

  const size_t SNF = (size_t)BATCH * N_DIM;      // state elements (256K)
  char* ws = (char*)d_ws;
  float*    x    = (float*)   (ws);                                   // 1 MB
  float*    y    = (float*)   (ws + SNF * 4);                         // 1 MB
  _Float16* y16a = (_Float16*)(ws + SNF * 8);                         // 0.5 MB
  _Float16* y16b = (_Float16*)(ws + SNF * 10);                        // 0.5 MB
  _Float16* h16  = (_Float16*)(ws + SNF * 12);                        // 2 MB

  horn_init<<<(BATCH * N_DIM + 255) / 256, 256, 0, stream>>>(x, y, y16a);
  horn_cvt_w<<<(N_DIM * N_DIM + 255) / 256, 256, 0, stream>>>(h2h_w, h16);

  for (int t = 0; t < T_STEPS; ++t) {
    const _Float16* yi = (t & 1) ? y16b : y16a;
    _Float16*       yo = (t & 1) ? y16a : y16b;
    horn_step<<<256, 128, 0, stream>>>(
        batch + (size_t)t * BATCH * I_DIM,
        i2h_w, i2h_b, h16, h2h_b, x, y, yi, yo);
  }

  horn_out<<<(BATCH * O_DIM + 255) / 256, 256, 0, stream>>>(x, h2o_w, h2o_b, out);
}
